// StochasticAttention_70437463654930
// MI455X (gfx1250) — matrix operand order repaired and checked by reading, not compiled, not run on hardware
//
#include <hip/hip_runtime.h>
#include <hip/hip_bf16.h>
#include <hip/hip_fp16.h>

typedef __attribute__((ext_vector_type(8)))  _Float16 v8h;
typedef __attribute__((ext_vector_type(16))) _Float16 v16h;
typedef __attribute__((ext_vector_type(8)))  float    v8f;

#define B_   64
#define N_   196
#define D_   768
#define H_   12
#define HD_  64
#define SIDE_ 14
#define NP_  224          // tokens padded per batch to multiple of 32
#define MT_  (NP_ / 16)   // 14 row tiles per batch

static __device__ __forceinline__ v8f wmma_f16(v16h a, v16h b, v8f c) {
    return __builtin_amdgcn_wmma_f32_16x16x32_f16(false, a, false, b,
                                                  (short)0, c, false, false);
}

// A-fragment (16x32 f16): lane = row; K = 2*klo .. 2*klo+15 (contiguous)
static __device__ __forceinline__ v16h load_afrag(const _Float16* __restrict__ row,
                                                  int kk, int klo) {
    v8h lo = *(const v8h*)(row + kk + 2 * klo);
    v8h hi = *(const v8h*)(row + kk + 2 * klo + 8);
    v16h r;
#pragma unroll
    for (int i = 0; i < 8; ++i) { r[i] = lo[i]; r[8 + i] = hi[i]; }
    return r;
}

// ---------------- conversion kernels ----------------
__global__ __launch_bounds__(256) void k_f32tof16(const float* __restrict__ src,
                                                  _Float16* __restrict__ dst, int n) {
    int i = blockIdx.x * 256 + threadIdx.x;
    if (i < n) dst[i] = (_Float16)src[i];
}

__global__ __launch_bounds__(256) void k_zero16(_Float16* __restrict__ dst, int n) {
    int i = blockIdx.x * 256 + threadIdx.x;
    if (i < n) dst[i] = (_Float16)0.f;
}

// WT[n][k] = W[k][n]  (768x768)
__global__ __launch_bounds__(256) void k_transpose16(const float* __restrict__ W,
                                                     _Float16* __restrict__ WT) {
    int i = blockIdx.x * 256 + threadIdx.x;
    if (i >= D_ * D_) return;
    int n = i / D_, k = i % D_;
    WT[i] = (_Float16)W[k * D_ + n];
}

// fully-unrolled 16x64 GEMM K-loop (shared by k_qkv / k_out).
// No loop-carried fragment registers: every load feeds its WMMA directly and
// the scheduler hoists loads ahead (copy-free software pipelining).
static __device__ __forceinline__ void gemm_k_loop(const _Float16* __restrict__ arow,
                                                   const _Float16* const colp[4],
                                                   int klo, v8f acc[4]) {
#pragma unroll
    for (int kk = 0; kk < D_; kk += 32) {
        v16h a = load_afrag(arow, kk, klo);
#pragma unroll
        for (int t = 0; t < 4; ++t) {
            v16h bfrag = *(const v16h*)(colp[t] + kk);
            acc[t] = wmma_f16(a, bfrag, acc[t]);
        }
    }
}

// ---------------- fused QKV projection ----------------
// Q,K: [B][H][NP_][HD_] f16 ; Vt: [B][H][HD_][NP_] f16 (transposed)
__global__ __launch_bounds__(256) void k_qkv(const _Float16* __restrict__ hh,
                                             const _Float16* __restrict__ zrow,
                                             const _Float16* __restrict__ WT3,
                                             const float* __restrict__ bq,
                                             const float* __restrict__ bk,
                                             const float* __restrict__ bv,
                                             _Float16* __restrict__ Q,
                                             _Float16* __restrict__ K,
                                             _Float16* __restrict__ Vt) {
    const int lane = threadIdx.x & 31;
    int wid = blockIdx.x * 8 + (threadIdx.x >> 5);
    const int ng = wid % (D_ / 64); wid /= (D_ / 64);   // head (64-wide col group)
    const int mt = wid % MT_;       wid /= MT_;         // 16-row tile in batch
    const int w  = wid % 3;         wid /= 3;           // 0=Q 1=K 2=V
    const int b  = wid;

    const int col = lane & 15;
    const int klo = (lane < 16) ? 0 : 8;
    const int kb  = (lane < 16) ? 0 : 16;
    const int mb  = (lane < 16) ? 0 : 8;

    const int token = mt * 16 + col;       // A-fragment row
    const bool vrow = token < N_;
    // invalid rows read from a zero-filled row: branch-free inner loop
    const _Float16* arow = vrow ? (hh + (size_t)(b * N_ + token) * D_) : zrow;
    const _Float16* WT   = WT3 + (size_t)w * D_ * D_;

    const int n0 = ng * 64 + col;
    const _Float16* colp[4];
#pragma unroll
    for (int t = 0; t < 4; ++t)
        colp[t] = WT + (size_t)(n0 + t * 16) * D_ + kb;

    v8f acc[4] = {};
    gemm_k_loop(arow, colp, klo, acc);

    if (w < 2) {
        _Float16* dst = (w == 0) ? Q : K;
        const float* bias = (w == 0) ? bq : bk;
#pragma unroll
        for (int t = 0; t < 4; ++t) {
            int hd = t * 16 + col;
            float bsv = bias[ng * 64 + hd];
            size_t base = (((size_t)b * H_ + ng) * NP_) * HD_ + hd;
#pragma unroll
            for (int j = 0; j < 8; ++j) {
                int tok = mt * 16 + mb + j;
                float vvl = (tok < N_) ? (acc[t][j] + bsv) : 0.f;
                dst[base + (size_t)tok * HD_] = (_Float16)vvl;
            }
        }
    } else {
#pragma unroll
        for (int t = 0; t < 4; ++t) {
            int hd = t * 16 + col;
            float bsv = bv[ng * 64 + hd];
            v8h pack;
#pragma unroll
            for (int j = 0; j < 8; ++j) {
                int tok = mt * 16 + mb + j;
                pack[j] = (tok < N_) ? (_Float16)(acc[t][j] + bsv) : (_Float16)0.f;
            }
            size_t off = (((size_t)b * H_ + ng) * HD_ + hd) * NP_ + mt * 16 + mb;
            *(v8h*)(Vt + off) = pack;
        }
    }
}

// ---------------- per-query sigma -> 1/denom ----------------
__global__ __launch_bounds__(256) void k_sigma(const _Float16* __restrict__ Q,
                                               const float* __restrict__ Wsig,
                                               const float* __restrict__ bsig,
                                               const float* __restrict__ temp,
                                               float* __restrict__ invden) {
    int i = blockIdx.x * 256 + threadIdx.x;
    if (i >= B_ * H_ * N_) return;
    int q = i % N_;
    int bh = i / N_;
    const _Float16* qp = Q + ((size_t)bh * NP_ + q) * HD_;
    float acc = 0.f;
#pragma unroll
    for (int d0 = 0; d0 < HD_; d0 += 8) {
        v8h qv = *(const v8h*)(qp + d0);
#pragma unroll
        for (int t = 0; t < 8; ++t) acc += (float)qv[t] * Wsig[d0 + t];
    }
    float x = acc + bsig[0];
    float sp = (x > 20.f) ? x : log1pf(__expf(x));
    float sig = sp + 1e-3f;
    float tt = temp[0];
    float den = 2.f * sig * sig * tt * tt;
    invden[i] = 1.f / den;
}

// ---------------- flash attention per (b,h) ----------------
__global__ __launch_bounds__(256) void k_attn(const _Float16* __restrict__ Q,
                                              const _Float16* __restrict__ K,
                                              const _Float16* __restrict__ Vt,
                                              const float* __restrict__ invden,
                                              _Float16* __restrict__ ctxh) {
    __shared__ alignas(16) _Float16 P[8][16][40];
    const int lane = threadIdx.x & 31;
    const int wv = threadIdx.x >> 5;
    const int bh = blockIdx.x;
    const int b = bh / H_, h = bh % H_;
    const _Float16* Qb = Q + (size_t)bh * NP_ * HD_;
    const _Float16* Kb = K + (size_t)bh * NP_ * HD_;
    const _Float16* Vb = Vt + (size_t)bh * HD_ * NP_;
    const float* idb = invden + (size_t)bh * N_;

    const int col = lane & 15;
    const int mb = (lane < 16) ? 0 : 8;
    const int klo = (lane < 16) ? 0 : 8;
    const int kb = (lane < 16) ? 0 : 16;

    for (int qt = wv; qt < MT_; qt += 8) {
        // Q A-fragments (K = 0..31, 32..63)
        const _Float16* qp = Qb + (size_t)(qt * 16 + col) * HD_;
        v16h aq0 = {}, aq1 = {};
        {
            v8h a = *(const v8h*)(qp + 2 * klo);
            v8h bf = *(const v8h*)(qp + 2 * klo + 8);
            v8h c = *(const v8h*)(qp + 32 + 2 * klo);
            v8h d = *(const v8h*)(qp + 32 + 2 * klo + 8);
#pragma unroll
            for (int i = 0; i < 8; ++i) {
                aq0[i] = a[i]; aq0[8 + i] = bf[i];
                aq1[i] = c[i]; aq1[8 + i] = d[i];
            }
        }
        float idr[8]; int qy[8], qx[8];
#pragma unroll
        for (int j = 0; j < 8; ++j) {
            int tq = qt * 16 + mb + j;
            idr[j] = (tq < N_) ? idb[tq] : 0.f;
            qy[j] = tq / SIDE_; qx[j] = tq % SIDE_;
        }
        float rmax[8], rsum[8];
        v8f acc[4] = {};
#pragma unroll
        for (int j = 0; j < 8; ++j) { rmax[j] = -1e30f; rsum[j] = 0.f; }

        for (int kp = 0; kp < NP_ / 32; ++kp) {
            const int key0 = kp * 32 + col;
            const int key1 = key0 + 16;
            const _Float16* kp0 = Kb + (size_t)key0 * HD_;
            const _Float16* kp1 = Kb + (size_t)key1 * HD_;
            // issue all global loads for this step up front: K fragments feed the
            // score WMMAs; V fragments are consumed only after softmax + LDS
            // transpose, so their latency hides under that work.
            v16h bk00 = *(const v16h*)(kp0 + kb);
            v16h bk01 = *(const v16h*)(kp0 + 32 + kb);
            v16h bk10 = *(const v16h*)(kp1 + kb);
            v16h bk11 = *(const v16h*)(kp1 + 32 + kb);
            v16h bvf[4];
#pragma unroll
            for (int f = 0; f < 4; ++f)
                bvf[f] = *(const v16h*)(Vb + (size_t)(f * 16 + col) * NP_ + kp * 32 + kb);

            v8f s0 = {}, s1 = {};
            s0 = wmma_f16(aq0, bk00, s0);
            s0 = wmma_f16(aq1, bk01, s0);
            s1 = wmma_f16(aq0, bk10, s1);
            s1 = wmma_f16(aq1, bk11, s1);

            const bool kv0 = key0 < N_, kv1 = key1 < N_;
            const int k0y = key0 / SIDE_, k0x = key0 % SIDE_;
            const int k1y = key1 / SIDE_, k1x = key1 % SIDE_;
            float p0[8], p1[8], tmax[8];
#pragma unroll
            for (int j = 0; j < 8; ++j) {
                float dy0 = (float)(qy[j] - k0y), dx0 = (float)(qx[j] - k0x);
                float dy1 = (float)(qy[j] - k1y), dx1 = (float)(qx[j] - k1x);
                float sv0 = kv0 ? s0[j] * 0.125f * __expf(-(dy0 * dy0 + dx0 * dx0) * idr[j]) : -1e30f;
                float sv1 = kv1 ? s1[j] * 0.125f * __expf(-(dy1 * dy1 + dx1 * dx1) * idr[j]) : -1e30f;
                p0[j] = sv0; p1[j] = sv1;
                tmax[j] = fmaxf(sv0, sv1);
            }
#pragma unroll
            for (int j = 0; j < 8; ++j) {
#pragma unroll
                for (int o = 1; o < 16; o <<= 1)
                    tmax[j] = fmaxf(tmax[j], __shfl_xor(tmax[j], o, 32));
            }
#pragma unroll
            for (int j = 0; j < 8; ++j) {
                float mn = fmaxf(rmax[j], tmax[j]);
                float sc = __expf(rmax[j] - mn);
                rmax[j] = mn;
                p0[j] = __expf(p0[j] - mn);
                p1[j] = __expf(p1[j] - mn);
                rsum[j] = rsum[j] * sc + p0[j] + p1[j];
#pragma unroll
                for (int f = 0; f < 4; ++f) acc[f][j] *= sc;
            }
            // transpose P tile through LDS (wave-private region; DS ops in-order)
            __builtin_amdgcn_wave_barrier();
#pragma unroll
            for (int j = 0; j < 8; ++j) {
                P[wv][mb + j][col] = (_Float16)p0[j];
                P[wv][mb + j][col + 16] = (_Float16)p1[j];
            }
            __builtin_amdgcn_wave_barrier();
            v16h ap = {};
            {
                const _Float16* pr = &P[wv][col][0];
                v8h plo = *(const v8h*)(pr + 2 * klo);
                v8h phi = *(const v8h*)(pr + 2 * klo + 8);
#pragma unroll
                for (int i = 0; i < 8; ++i) { ap[i] = plo[i]; ap[8 + i] = phi[i]; }
            }
#pragma unroll
            for (int f = 0; f < 4; ++f)
                acc[f] = wmma_f16(ap, bvf[f], acc[f]);
        }
        // finalize: reduce row sums across the 16 lanes, normalize, store
#pragma unroll
        for (int j = 0; j < 8; ++j) {
#pragma unroll
            for (int o = 1; o < 16; o <<= 1)
                rsum[j] += __shfl_xor(rsum[j], o, 32);
            rsum[j] = 1.f / rsum[j];
        }
#pragma unroll
        for (int f = 0; f < 4; ++f) {
#pragma unroll
            for (int j = 0; j < 8; ++j) {
                int tq = qt * 16 + mb + j;
                if (tq < N_)
                    ctxh[(size_t)(b * N_ + tq) * D_ + h * HD_ + f * 16 + col] =
                        (_Float16)(acc[f][j] * rsum[j]);
            }
        }
    }
}

// ---------------- output projection ----------------
__global__ __launch_bounds__(256) void k_out(const _Float16* __restrict__ ctx,
                                             const _Float16* __restrict__ WoT,
                                             const float* __restrict__ bo,
                                             float* __restrict__ out) {
    const int lane = threadIdx.x & 31;
    int wid = blockIdx.x * 8 + (threadIdx.x >> 5);
    const int ng = wid % (D_ / 64); wid /= (D_ / 64);
    const int mt = wid;                         // 0..783
    const int col = lane & 15;
    const int klo = (lane < 16) ? 0 : 8;
    const int kb = (lane < 16) ? 0 : 16;
    const int mb = (lane < 16) ? 0 : 8;

    const _Float16* arow = ctx + (size_t)(mt * 16 + col) * D_;
    const int n0 = ng * 64 + col;
    const _Float16* colp[4];
#pragma unroll
    for (int t = 0; t < 4; ++t)
        colp[t] = WoT + (size_t)(n0 + t * 16) * D_ + kb;

    v8f acc[4] = {};
    gemm_k_loop(arow, colp, klo, acc);

#pragma unroll
    for (int t = 0; t < 4; ++t) {
        int n = ng * 64 + t * 16 + col;
        float bsv = bo[n];
#pragma unroll
        for (int j = 0; j < 8; ++j) {
            int row = mt * 16 + mb + j;
            out[(size_t)row * D_ + n] = acc[t][j] + bsv;
        }
    }
}

extern "C" void kernel_launch(void* const* d_in, const int* in_sizes, int n_in,
                              void* d_out, int out_size, void* d_ws, size_t ws_size,
                              hipStream_t stream) {
    (void)in_sizes; (void)n_in; (void)out_size; (void)ws_size;
    const float* hs   = (const float*)d_in[0];
    const float* temp = (const float*)d_in[1];
    const float* Wq   = (const float*)d_in[2];
    const float* bq   = (const float*)d_in[3];
    const float* Wk   = (const float*)d_in[4];
    const float* bk   = (const float*)d_in[5];
    const float* Wv   = (const float*)d_in[6];
    const float* bv   = (const float*)d_in[7];
    const float* Wo   = (const float*)d_in[8];
    const float* bo   = (const float*)d_in[9];
    const float* Wsig = (const float*)d_in[10];
    const float* bsig = (const float*)d_in[11];
    float* out = (float*)d_out;

    char* ws = (char*)d_ws;
    size_t off = 0;
    auto take = [&](size_t bytes) -> char* {
        char* p = ws + off;
        off += (bytes + 255) & ~(size_t)255;
        return p;
    };
    _Float16* hh   = (_Float16*)take((size_t)B_ * N_ * D_ * 2);
    _Float16* zrow = (_Float16*)take((size_t)D_ * 2);
    _Float16* WT3  = (_Float16*)take((size_t)3 * D_ * D_ * 2);
    _Float16* WoT  = (_Float16*)take((size_t)D_ * D_ * 2);
    _Float16* Qh   = (_Float16*)take((size_t)B_ * H_ * NP_ * HD_ * 2);
    _Float16* Kh   = (_Float16*)take((size_t)B_ * H_ * NP_ * HD_ * 2);
    _Float16* Vt   = (_Float16*)take((size_t)B_ * H_ * HD_ * NP_ * 2);
    _Float16* ctxh = (_Float16*)take((size_t)B_ * N_ * D_ * 2);
    float*    invd = (float*)take((size_t)B_ * H_ * N_ * 4);

    const int nh = B_ * N_ * D_;
    k_f32tof16<<<(nh + 255) / 256, 256, 0, stream>>>(hs, hh, nh);
    k_zero16<<<(D_ + 255) / 256, 256, 0, stream>>>(zrow, D_);
    const int nw = D_ * D_;
    k_transpose16<<<(nw + 255) / 256, 256, 0, stream>>>(Wq, WT3);
    k_transpose16<<<(nw + 255) / 256, 256, 0, stream>>>(Wk, WT3 + (size_t)D_ * D_);
    k_transpose16<<<(nw + 255) / 256, 256, 0, stream>>>(Wv, WT3 + (size_t)2 * D_ * D_);
    k_transpose16<<<(nw + 255) / 256, 256, 0, stream>>>(Wo, WoT);

    // B*3*14*12 = 32256 waves -> 4032 blocks of 8 waves
    k_qkv<<<4032, 256, 0, stream>>>(hh, zrow, WT3, bq, bk, bv, Qh, Kh, Vt);
    k_sigma<<<(B_ * H_ * N_ + 255) / 256, 256, 0, stream>>>(Qh, Wsig, bsig, temp, invd);
    k_attn<<<B_ * H_, 256, 0, stream>>>(Qh, Kh, Vt, invd, ctxh);
    // 784*12 = 9408 waves -> 1176 blocks
    k_out<<<1176, 256, 0, stream>>>(ctxh, WoT, bo, out);
}
